// RLSTMModel_53128745452232
// MI455X (gfx1250) — matrix operand order repaired and checked by reading, not compiled, not run on hardware
//
#include <hip/hip_runtime.h>
#include <hip/hip_bf16.h>
#include <math.h>

// ---------------------------------------------------------------------------
// rLSTM (B=128, T=1024, H=64, MI=1, AI=32, D=97) for gfx1250 / MI455X.
//
// 4 workgroups (1 per WGP), each owning 32 batch rows (2 WMMA M-fragments)
// and running the sequential T-loop privately. Per step each WG does a
// [32 x 128] x [128 x 4288] bf16 WMMA GEMM (K padded 97->128). Each B
// fragment feeds 2 WMMAs. The first 60 n-tiles (240 KB) of the prepacked
// weights are cached permanently in LDS (loaded once, reused for all 1024
// steps); the rest streams from the 192 MB L2. 24 wave32 waves per WG
// (6/SIMD) balance latency hiding against the VGPR budget. The 8 A fragments
// are explicitly loaded into registers once per step (before the N sweep) so
// the LDS atomics in the epilogue cannot force per-tile re-loads. The c . r
// einsum is fused into the GEMM epilogue with ds_add_f32 atomics; c_old is
// kept transposed ([h][m]) so per-tile broadcast reads are two ds_load_b128.
// ---------------------------------------------------------------------------

typedef __attribute__((ext_vector_type(16))) __bf16 v16bf;
typedef __attribute__((ext_vector_type(8)))  float  v8f;

constexpr int kB   = 128;
constexpr int kT   = 1024;
constexpr int kH   = 64;
constexpr int kMI  = 1;
constexpr int kAI  = 32;
constexpr int kD   = kMI + kAI + kH;   // 97
constexpr int kNG  = 2 * kH;           // 128 (gates)
constexpr int kNC  = kH;               // 64  (conn)
constexpr int kNR  = kH * kH;          // 4096 (redistribution)
constexpr int kNTOT = kNG + kNC + kNR; // 4288
constexpr int kNT  = kNTOT / 16;       // 268 n-tiles of width 16
constexpr int kKT  = 4;                // 4 k-tiles of 32 (K padded to 128)
constexpr int kMTILE = 32;             // batch rows per workgroup (2 m-frags)
constexpr int kWG    = kB / kMTILE;    // 4 workgroups
constexpr int kWAVES = 24;             // wave32 waves per workgroup (6/SIMD)
constexpr int kTHREADS = kWAVES * 32;  // 768
constexpr int kTileElems = kKT * 32 * 16; // 2048 bf16 per n-tile (4 KB)
constexpr int kRES = 60;               // n-tiles resident in LDS (240 KB)

// A-fragment slot for element (k, m) of a 16x32 bf16 A matrix, per the CDNA5
// ISA 16-bit A layout: lanes 0-15 hold K={0..7,16..23}, lanes 16-31 hold
// K={8..15,24..31}; within a lane the 16 halfs are VGPR-ordered.
__device__ __forceinline__ void put_x(__bf16 (&xf)[kKT][32][16], int k, int m,
                                      __bf16 v) {
  const int kt   = k >> 5;
  const int kk   = k & 31;
  const int lane = (((kk >> 3) & 1) << 4) | m;
  const int j    = (kk & 16) ? (8 + (kk & 7)) : (kk & 7);
  xf[kt][lane][j] = v;
}

// --------------------------- weight prepack --------------------------------
// Packs [W_gates | W_conn | W_r] (concat along N, K zero-padded to 128) into
// bf16 WMMA B-fragments: for n-tile nt, k-tile kt, lane L holds the 16
// K values (j/8)*16 + (L/16)*8 + (j%8) of column nt*16 + (L%16).
__global__ void rlstm_prepack_kernel(const float* __restrict__ Wg,
                                     const float* __restrict__ Wc,
                                     const float* __restrict__ Wr,
                                     __bf16* __restrict__ outp) {
  const int idx = blockIdx.x * blockDim.x + threadIdx.x;
  if (idx >= kNT * kTileElems) return;
  const int j    = idx & 15;
  const int lane = (idx >> 4) & 31;
  const int kt   = (idx >> 9) & 3;
  const int nt   = idx >> 11;
  const int k = kt * 32 + ((j >> 3) << 4) + ((lane >> 4) << 3) + (j & 7);
  const int n = nt * 16 + (lane & 15);
  float v = 0.0f;
  if (k < kD) {
    if (n < kNG)            v = Wg[k * kNG + n];
    else if (n < kNG + kNC) v = Wc[k * kNC + (n - kNG)];
    else                    v = Wr[k * kNR + (n - kNG - kNC)];
  }
  outp[idx] = (__bf16)v;
}

// ------------------------------ main kernel --------------------------------
__global__ __launch_bounds__(kTHREADS)
void rlstm_scan_kernel(const float* __restrict__ x_m,
                       const float* __restrict__ x_a,
                       const float* __restrict__ b_gates,
                       const float* __restrict__ b_conn,
                       const float* __restrict__ b_r,
                       const float* __restrict__ W_fc,
                       const float* __restrict__ b_fc,
                       const __bf16* __restrict__ Wpack,
                       float* __restrict__ out) {
  // Permanent LDS weight cache: first kRES n-tiles (240 KB).
  __shared__ alignas(16) __bf16 wres[kRES * kTileElems];
  __shared__ __bf16 xfrag[2][kKT][32][16];  // A fragments, 2 m-frags (8 KB)
  __shared__ float c_oldT[kH][kMTILE];      // c transposed [h][m] (8 KB)
  __shared__ float c_dot[kMTILE][kH];       // fused einsum accumulator (8 KB)
  __shared__ float g_raw[kMTILE][kNG];      // (16 KB)
  __shared__ float s_raw[kMTILE][kH];       // (8 KB)
  __shared__ float h_st[kMTILE][kH];        // (8 KB)

  const int tid  = threadIdx.x;
  const int wave = tid >> 5;
  const int lane = tid & 31;
  const int m0   = blockIdx.x * kMTILE;     // first batch row of this WG

  // One-time: fill the LDS weight cache (16 B chunks, fully coalesced).
  for (int i = tid; i < kRES * kTileElems / 8; i += kTHREADS)
    *(float4*)&wres[i * 8] = *(const float4*)&Wpack[i * 8];

  // Init: zero x fragments (covers K pad 97..127 and the h slots), c0, h0.
  for (int i = tid; i < 2 * kKT * 32 * 16; i += kTHREADS)
    ((__bf16*)xfrag)[i] = (__bf16)0.0f;
  for (int i = tid; i < kMTILE * kH; i += kTHREADS) {
    ((float*)c_oldT)[i] = 0.0f;
    ((float*)h_st)[i]   = 0.0f;
  }
  __syncthreads();

  for (int t = 0; t < kT; ++t) {
    // ---- Phase A: zero c_dot, stage x_m / x_a into the A fragments -------
    for (int i = tid; i < kMTILE * kH; i += kTHREADS) ((float*)c_dot)[i] = 0.0f;
    for (int i = tid; i < kMTILE * (kMI + kAI); i += kTHREADS) {
      const int m = i / (kMI + kAI);
      const int f = i % (kMI + kAI);
      const int bg = m0 + m;
      const float v = (f == 0) ? x_m[bg * kT + t]
                               : x_a[(bg * kT + t) * kAI + (f - 1)];
      put_x(xfrag[m >> 4], f, m & 15, (__bf16)v);
    }
    __syncthreads();

    // ---- Phase B: [32x128]x[128x4288] bf16 WMMA GEMM + fused epilogue ----
    // Explicitly pin all 8 A fragments in registers for the whole N sweep so
    // the epilogue's LDS atomics cannot force per-tile re-loads.
    v16bf afr[2][kKT];
#pragma unroll
    for (int mf = 0; mf < 2; ++mf)
#pragma unroll
      for (int kt = 0; kt < kKT; ++kt)
        afr[mf][kt] = *(const v16bf*)(&xfrag[mf][kt][lane][0]);

    for (int nt = wave; nt < kNT; nt += kWAVES) {
      v8f acc0 = {0.f, 0.f, 0.f, 0.f, 0.f, 0.f, 0.f, 0.f};
      v8f acc1 = acc0;
      if (nt < kRES) {
        // LDS-resident weight tiles: ds_load_b128 path.
        const __bf16* lp = &wres[nt * kTileElems + lane * 16];
#pragma unroll
        for (int kt = 0; kt < kKT; ++kt) {
          v16bf b = *(const v16bf*)(lp + kt * 512);
          acc0 = __builtin_amdgcn_wmma_f32_16x16x32_bf16(
              false, afr[0][kt], false, b, (short)0, acc0, false, false);
          acc1 = __builtin_amdgcn_wmma_f32_16x16x32_bf16(
              false, afr[1][kt], false, b, (short)0, acc1, false, false);
        }
      } else {
        // L2-streamed weight tiles; prefetch this wave's next tile.
        const __bf16* gp = Wpack + (size_t)nt * kTileElems + lane * 16;
        __builtin_prefetch(gp + (size_t)kWAVES * kTileElems, 0, 1);
#pragma unroll
        for (int kt = 0; kt < kKT; ++kt) {
          v16bf b = *(const v16bf*)(gp + kt * 512);
          acc0 = __builtin_amdgcn_wmma_f32_16x16x32_bf16(
              false, afr[0][kt], false, b, (short)0, acc0, false, false);
          acc1 = __builtin_amdgcn_wmma_f32_16x16x32_bf16(
              false, afr[1][kt], false, b, (short)0, acc1, false, false);
        }
      }
      // C/D fragment layout: VGPR r, lanes 0-15 -> M=r, lanes 16-31 -> M=8+r.
      const int n0 = nt * 16;
      const int n  = n0 + (lane & 15);
      const int mo = (lane >> 4) << 3;
      if (n0 < kNG) {                      // gates region
#pragma unroll
        for (int r = 0; r < 8; ++r) {
          g_raw[mo + r][n]      = acc0[r];
          g_raw[16 + mo + r][n] = acc1[r];
        }
      } else if (n0 < kNG + kNC) {         // conn region
#pragma unroll
        for (int r = 0; r < 8; ++r) {
          s_raw[mo + r][n - kNG]      = acc0[r];
          s_raw[16 + mo + r][n - kNG] = acc1[r];
        }
      } else {                             // r region: fuse c . r einsum
        const int col = n - (kNG + kNC);   // h*64 + k
        const int hi  = col >> 6;          // uniform within each lane-half
        const int kk  = col & 63;
        const float br = b_r[col];
        // Broadcast c_old rows: 8 contiguous floats per m-frag -> 2x b128.
        float cv0[8], cv1[8];
        *(float4*)&cv0[0] = *(const float4*)&c_oldT[hi][mo];
        *(float4*)&cv0[4] = *(const float4*)&c_oldT[hi][mo + 4];
#pragma unroll
        for (int r = 0; r < 8; ++r)
          atomicAdd(&c_dot[mo + r][kk], cv0[r] * (acc0[r] + br));
        *(float4*)&cv1[0] = *(const float4*)&c_oldT[hi][16 + mo];
        *(float4*)&cv1[4] = *(const float4*)&c_oldT[hi][16 + mo + 4];
#pragma unroll
        for (int r = 0; r < 8; ++r)
          atomicAdd(&c_dot[16 + mo + r][kk], cv1[r] * (acc1[r] + br));
      }
    }
    __syncthreads();

    // ---- Phase C: nonlinearity + state update ----------------------------
    for (int i = tid; i < kMTILE * kH; i += kTHREADS) {
      const int m = i >> 6;
      const int k = i & 63;
      const float ig = 1.0f / (1.0f + __expf(-(g_raw[m][k] + b_gates[k])));
      const float og = 1.0f / (1.0f + __expf(-(g_raw[m][kH + k] + b_gates[kH + k])));
      const float s  = s_raw[m][k] + b_conn[k];
      const float c  = c_dot[m][k] + ig * tanhf(s);
      const float h  = og * tanhf(c);
      c_oldT[k][m] = c;
      h_st[m][k]   = h;
      put_x(xfrag[m >> 4], kMI + kAI + k, m & 15, (__bf16)h);
    }
    __syncthreads();
  }

  // ---- Outputs: pred = h_T @ W_fc + b_fc ; state = [c_T ; h_T] -----------
  for (int i = tid; i < kMTILE; i += kTHREADS) {
    float acc = b_fc[0];
#pragma unroll 8
    for (int k = 0; k < kH; ++k) acc += h_st[i][k] * W_fc[k];
    out[m0 + i] = acc;
  }
  for (int i = tid; i < kMTILE * kH; i += kTHREADS) {
    const int m = i >> 6;
    const int k = i & 63;
    out[kB + (size_t)(m0 + m) * kH + k]                   = c_oldT[k][m];
    out[kB + (size_t)kB * kH + (size_t)(m0 + m) * kH + k] = h_st[m][k];
  }
}

// ---------------------------------------------------------------------------
extern "C" void kernel_launch(void* const* d_in, const int* in_sizes, int n_in,
                              void* d_out, int out_size, void* d_ws,
                              size_t ws_size, hipStream_t stream) {
  (void)in_sizes; (void)n_in; (void)out_size; (void)ws_size;
  const float* x_m     = (const float*)d_in[0];
  const float* x_a     = (const float*)d_in[1];
  const float* W_gates = (const float*)d_in[2];
  const float* b_gates = (const float*)d_in[3];
  const float* W_conn  = (const float*)d_in[4];
  const float* b_conn  = (const float*)d_in[5];
  const float* W_r     = (const float*)d_in[6];
  const float* b_r     = (const float*)d_in[7];
  const float* W_fc    = (const float*)d_in[8];
  const float* b_fc    = (const float*)d_in[9];
  __bf16* Wpack = (__bf16*)d_ws;  // 268*2048 bf16 = ~1.07 MB

  const int total = kNT * kTileElems;
  rlstm_prepack_kernel<<<(total + 255) / 256, 256, 0, stream>>>(
      W_gates, W_conn, W_r, Wpack);
  rlstm_scan_kernel<<<kWG, kTHREADS, 0, stream>>>(
      x_m, x_a, b_gates, b_conn, b_r, W_fc, b_fc, Wpack, (float*)d_out);
}
